// MultiQueryAttention_3418793968324
// MI455X (gfx1250) — matrix operand order repaired and checked by reading, not compiled, not run on hardware
//
#include <hip/hip_runtime.h>
#include <math.h>

// ---------------------------------------------------------------------------
// MQA forward for MI455X (gfx1250, wave32, WMMA).
// f32->bf16 convert (+weight transpose) -> WMMA GEMMs for Q/K/V ->
// flash-attention computed TRANSPOSED (S^T = K Q^T, O^T = V^T P^T) so softmax
// state is per-lane scalar and cross-lane traffic is minimal -> WMMA out-proj.
// GEMM waves compute 32x64 strips: each weight B-fragment feeds 2 WMMAs.
// ---------------------------------------------------------------------------

typedef __bf16 bf16;
typedef bf16  v8bf  __attribute__((ext_vector_type(8)));
typedef bf16  v16bf __attribute__((ext_vector_type(16)));
typedef float v8f   __attribute__((ext_vector_type(8)));

#define NB    2
#define NSEQ  2048
#define EDIM  1024
#define NH    16
#define DH    64
#define HDIM  1024
#define VTLD  (NSEQ + 32)   // padded leading dim for transposed V
#define PLD   40            // LDS row stride (elements) for P: 16B-aligned, conflict-spreading

__device__ __forceinline__ v16bf cat16(v8bf lo, v8bf hi) {
  return __builtin_shufflevector(lo, hi, 0,1,2,3,4,5,6,7,8,9,10,11,12,13,14,15);
}

__device__ __forceinline__ v8f wmma_bf16(v16bf a, v16bf b, v8f c) {
  return __builtin_amdgcn_wmma_f32_16x16x32_bf16(false, a, false, b, (short)0, c,
                                                 false, false);
}

// ----------------------------- converters ----------------------------------

__global__ void k_cvt(const float* __restrict__ in, bf16* __restrict__ out, int n) {
  int i = blockIdx.x * blockDim.x + threadIdx.x;
  if (i < n) out[i] = (bf16)in[i];
}

// W is (K x Nc) row-major f32; write Wt (Nc x K) row-major bf16.
__global__ void k_cvt_tr(const float* __restrict__ W, bf16* __restrict__ Wt,
                         int K, int Nc) {
  int i = blockIdx.x * blockDim.x + threadIdx.x;
  if (i < K * Nc) {
    int n = i / K;
    int k = i - n * K;
    Wt[i] = (bf16)W[k * Nc + n];
  }
}

// ------------------------------- GEMM ---------------------------------------
// C[M,Nc] = A[M,K] @ B[K,Nc], with B supplied transposed (Bt is Nc x K).
// One wave computes a 32x64 strip (2 M-tiles x 4 N-tiles): per 32-wide k-step,
// 4 A-fragment loads + 8 B-fragment loads feed 8 WMMAs (B reused across the
// two M-tiles). MODE: 0 = bf16 out, 1 = bf16 transposed out (V: Vt[b][d][n],
// leading dim VTLD), 2 = f32 out + bias (final projection).

template <int MODE>
__global__ void __launch_bounds__(128)
k_gemm(const bf16* __restrict__ A, const bf16* __restrict__ Bt,
       void* __restrict__ Cout, const float* __restrict__ bias,
       int M, int Nc, int K) {
  (void)bias;
  const int lane  = threadIdx.x & 31;
  const int wave  = blockIdx.x * (blockDim.x >> 5) + (threadIdx.x >> 5);
  const int mtiles = M >> 5;                 // 32-row strips
  const int total  = mtiles * (Nc >> 6);
  if (wave >= total) return;                 // wave-uniform: EXEC stays full
  const int mt    = wave % mtiles;
  const int ns    = wave / mtiles;
  const int mbase = mt << 5;
  const int nbase = ns << 6;
  const int col   = lane & 15;
  const int hi    = lane >> 4;
  const int koffA = hi * 8;
  const int koffB = hi * 8;

  v8f acc[2][4] = {};
  for (int k0 = 0; k0 < K; k0 += 32) {
    const bf16* ap0 = A + (size_t)(mbase + col) * K + k0 + koffA;
    const bf16* ap1 = A + (size_t)(mbase + 16 + col) * K + k0 + koffA;
    __builtin_prefetch(ap0 + 32, 0, 1);      // global_prefetch for next k-step
    v16bf af0 = cat16(*(const v8bf*)ap0, *(const v8bf*)(ap0 + 16));
    v16bf af1 = cat16(*(const v8bf*)ap1, *(const v8bf*)(ap1 + 16));
#pragma unroll
    for (int t = 0; t < 4; ++t) {
      const bf16* bp = Bt + (size_t)(nbase + t * 16 + col) * K + k0 + koffB;
      v16bf bf = cat16(*(const v8bf*)bp, *(const v8bf*)(bp + 16));
      acc[0][t] = wmma_bf16(af0, bf, acc[0][t]);
      acc[1][t] = wmma_bf16(af1, bf, acc[1][t]);
    }
  }

#pragma unroll
  for (int u = 0; u < 2; ++u) {
#pragma unroll
    for (int t = 0; t < 4; ++t) {
#pragma unroll
      for (int r = 0; r < 8; ++r) {
        const int orow = mbase + u * 16 + r + hi * 8;
        const int ocol = nbase + t * 16 + col;
        const float v = acc[u][t][r];
        if (MODE == 0) {
          ((bf16*)Cout)[(size_t)orow * Nc + ocol] = (bf16)v;
        } else if (MODE == 1) {
          const int bb = orow / NSEQ;
          const int nn = orow - bb * NSEQ;
          ((bf16*)Cout)[((size_t)bb * Nc + ocol) * VTLD + nn] = (bf16)v;
        } else {
          ((float*)Cout)[(size_t)orow * Nc + ocol] = v + bias[ocol];
        }
      }
    }
  }
}

// --------------------------- flash attention --------------------------------
// One wave per (b, h, 16-query tile). All tiles computed transposed:
//   S^T[key][query] = K_tile @ Q^T   (C-layout: regs = keys, lanes = queries)
//   O^T[d][query]  += V^T_tile @ P^T
// so m/l/corr are per-lane scalars; only 2 cross-lane ops per 32-key step.

__global__ void __launch_bounds__(128)
k_attn(const bf16* __restrict__ Q, const bf16* __restrict__ Kb,
       const bf16* __restrict__ Vt, bf16* __restrict__ Z) {
  __shared__ bf16 psh[4][16 * PLD];

  const int lane = threadIdx.x & 31;
  const int wib  = threadIdx.x >> 5;
  const int w    = blockIdx.x * 4 + wib;
  const int qt   = w & 127;              // N/16 = 128 query tiles
  const int h    = (w >> 7) & 15;
  const int b    = w >> 11;
  const int qbase = qt << 4;
  const int col  = lane & 15;
  const int hi   = lane >> 4;
  const int koffA = hi * 8;
  const int koffB = hi * 8;
  const int q    = qbase + col;          // this lane's query (dup in hi half)

  // Q^T B-fragments: bq[c][j] = Q[q][h*64 + c*32 + koffB + j + (j>=8 ? 8 : 0)], two 8-runs.
  v16bf bq[2];
#pragma unroll
  for (int c = 0; c < 2; ++c) {
    const bf16* qp = Q + (size_t)(b * NSEQ + q) * HDIM + h * DH + c * 32 + koffB;
    bq[c] = cat16(*(const v8bf*)qp, *(const v8bf*)(qp + 16));
  }

  v8f o[4] = {};                         // O^T tiles: regs = d, lanes = query
  float m = -__builtin_inff();
  float l = 0.0f;

  bf16* pw = psh[wib];                   // P stored [query][key], stride PLD
  const float scale = 0.125f;            // 1/sqrt(64)
  const int qend = qbase + 15;

  for (int kb = 0; kb <= qend; kb += 32) {
    // ---- S^T = K_tile @ Q^T : two 16-key tiles, accum over d chunks ----
    v8f st0 = {}, st1 = {};
#pragma unroll
    for (int c = 0; c < 2; ++c) {
      int krow0 = kb + col;        if (krow0 > NSEQ - 1) krow0 = NSEQ - 1;
      int krow1 = kb + 16 + col;   if (krow1 > NSEQ - 1) krow1 = NSEQ - 1;
      const bf16* kp0 = Kb + (size_t)(b * NSEQ + krow0) * DH + c * 32 + koffA;
      const bf16* kp1 = Kb + (size_t)(b * NSEQ + krow1) * DH + c * 32 + koffA;
      v16bf a0 = cat16(*(const v8bf*)kp0, *(const v8bf*)(kp0 + 16));
      v16bf a1 = cat16(*(const v8bf*)kp1, *(const v8bf*)(kp1 + 16));
      st0 = wmma_bf16(a0, bq[c], st0);
      st1 = wmma_bf16(a1, bq[c], st1);
    }

    // ---- per-lane online softmax over 32 keys held in registers ----
    const int khi = kb + hi * 8;         // key of reg r (tile0) = khi + r
    float v0[8], v1[8];
#pragma unroll
    for (int r = 0; r < 8; ++r) {
      v0[r] = (khi + r      > q) ? -__builtin_inff() : st0[r] * scale;
      v1[r] = (khi + r + 16 > q) ? -__builtin_inff() : st1[r] * scale;
    }
    float tmax = fmaxf(v0[0], v1[0]);
#pragma unroll
    for (int r = 1; r < 8; ++r) tmax = fmaxf(tmax, fmaxf(v0[r], v1[r]));
    tmax = fmaxf(tmax, __shfl_xor(tmax, 16, 32));   // merge half-waves
    const float mnew = fmaxf(m, tmax);
    const float corr = __expf(m - mnew);
    m = mnew;

    float p0[8], p1[8];
    float rs = 0.0f;
#pragma unroll
    for (int r = 0; r < 8; ++r) {
      p0[r] = __expf(v0[r] - mnew);
      p1[r] = __expf(v1[r] - mnew);
      rs += p0[r] + p1[r];
    }
    rs += __shfl_xor(rs, 16, 32);
    l = l * corr + rs;
#pragma unroll
    for (int t = 0; t < 4; ++t) o[t] *= corr;       // per-lane rescale

    // ---- store P to LDS as [query][key] (transpose during store) ----
#pragma unroll
    for (int r = 0; r < 8; ++r) {
      pw[col * PLD + hi * 8 + r]      = (bf16)p0[r];
      pw[col * PLD + 16 + hi * 8 + r] = (bf16)p1[r];
    }
    asm volatile("s_wait_dscnt 0" ::: "memory");    // per-wave DS is in-order

    // P^T B-fragment: pb[j] = P[q][koffB + j + (j>=8 ? 8 : 0)] -> 2x16B LDS reads
    const bf16* pr = pw + col * PLD + koffB;
    v16bf pb = cat16(*(const v8bf*)pr, *(const v8bf*)(pr + 16));

    // ---- O^T += V^T_tile @ P^T  (Vt[b][d][n] contiguous A-fragments) ----
#pragma unroll
    for (int t = 0; t < 4; ++t) {
      const bf16* vp = Vt + (size_t)(b * DH + t * 16 + col) * VTLD + kb + koffA;
      v16bf va = cat16(*(const v8bf*)vp, *(const v8bf*)(vp + 16));
      o[t] = wmma_bf16(va, pb, o[t]);
    }
  }

  // ---- normalize, pack 8 consecutive head-dims, one b128 store per tile ----
  const float inv = 1.0f / l;
  bf16* zp = Z + (size_t)(b * NSEQ + q) * HDIM + h * DH + hi * 8;
#pragma unroll
  for (int t = 0; t < 4; ++t) {
    v8bf zv;
#pragma unroll
    for (int r = 0; r < 8; ++r) zv[r] = (bf16)(o[t][r] * inv);
    *(v8bf*)(zp + t * 16) = zv;
  }
}

// ------------------------------ launcher ------------------------------------

extern "C" void kernel_launch(void* const* d_in, const int* in_sizes, int n_in,
                              void* d_out, int out_size, void* d_ws, size_t ws_size,
                              hipStream_t stream) {
  (void)in_sizes; (void)n_in; (void)out_size;
  const float* X  = (const float*)d_in[0];
  const float* Wq = (const float*)d_in[1];
  const float* Wk = (const float*)d_in[2];
  const float* Wv = (const float*)d_in[3];
  const float* Wo = (const float*)d_in[4];
  const float* bo = (const float*)d_in[5];
  float* out = (float*)d_out;

  const int M = NB * NSEQ;                     // 4096 rows
  char* p = (char*)d_ws;
  auto carve = [&](size_t bytes) {
    void* r = (void*)p;
    p += (bytes + 255) & ~(size_t)255;
    return r;
  };
  bf16* Xb  = (bf16*)carve((size_t)M * EDIM * 2);
  bf16* Wqt = (bf16*)carve((size_t)HDIM * EDIM * 2);
  bf16* Wkt = (bf16*)carve((size_t)DH * EDIM * 2);
  bf16* Wvt = (bf16*)carve((size_t)DH * EDIM * 2);
  bf16* Wot = (bf16*)carve((size_t)HDIM * HDIM * 2);
  bf16* Qb  = (bf16*)carve((size_t)M * HDIM * 2);
  bf16* Kbf = (bf16*)carve((size_t)M * DH * 2);
  bf16* Vtb = (bf16*)carve((size_t)NB * DH * VTLD * 2);
  bf16* Zb  = (bf16*)carve((size_t)M * HDIM * 2);
  if ((size_t)(p - (char*)d_ws) > ws_size) return;

  // 1) convert activations + transpose-convert weights to bf16
  {
    const int n = M * EDIM;
    k_cvt<<<(n + 255) / 256, 256, 0, stream>>>(X, Xb, n);
  }
  k_cvt_tr<<<(EDIM * HDIM) / 256, 256, 0, stream>>>(Wq, Wqt, EDIM, HDIM);
  k_cvt_tr<<<(EDIM * DH)   / 256, 256, 0, stream>>>(Wk, Wkt, EDIM, DH);
  k_cvt_tr<<<(EDIM * DH)   / 256, 256, 0, stream>>>(Wv, Wvt, EDIM, DH);
  k_cvt_tr<<<(HDIM * HDIM) / 256, 256, 0, stream>>>(Wo, Wot, HDIM, HDIM);

  // 2) projections: waves = (M/32)*(Nc/64); 4 waves per 128-thread block
  k_gemm<0><<<((M / 32) * (HDIM / 64)) / 4, 128, 0, stream>>>(
      Xb, Wqt, (void*)Qb, nullptr, M, HDIM, EDIM);
  k_gemm<0><<<((M / 32) * (DH / 64)) / 4, 128, 0, stream>>>(
      Xb, Wkt, (void*)Kbf, nullptr, M, DH, EDIM);
  k_gemm<1><<<((M / 32) * (DH / 64)) / 4, 128, 0, stream>>>(
      Xb, Wvt, (void*)Vtb, nullptr, M, DH, EDIM);

  // 3) causal MQA attention: waves = B*H*(N/16) = 4096
  k_attn<<<(NB * NH * (NSEQ / 16)) / 4, 128, 0, stream>>>(Qb, Kbf, Vtb, Zb);

  // 4) output projection with bias, f32 result
  k_gemm<2><<<((M / 32) * (HDIM / 64)) / 4, 128, 0, stream>>>(
      Zb, Wot, (void*)out, bo, M, HDIM, EDIM);
}
